// My_GRU_Cell_78580721648050
// MI455X (gfx1250) — matrix operand order repaired and checked by reading, not compiled, not run on hardware
//
#include <hip/hip_runtime.h>
#include <hip/hip_bf16.h>
#include <math.h>

// ---------------------------------------------------------------------------
// GRU (B=64, T=512, D=U=512) for gfx1250 (CDNA5, wave32, WMMA).
// Phase 1: x-projection for all t (parallel 51.5 GFLOP) via bf16 WMMA,
//          64x64 block tiles (2 WMMAs per wave per staged K-chunk).
// Phase 2: sequential h-recurrence via persistent kernel + grid sync,
//          bf16 WMMA for h@W, fp32 state and elementwise math.
// ---------------------------------------------------------------------------

typedef __bf16 bf16;
typedef __attribute__((ext_vector_type(8)))  bf16  bf16x8;
typedef __attribute__((ext_vector_type(16))) bf16  bf16x16;
typedef __attribute__((ext_vector_type(8)))  float f32x8;
typedef __attribute__((ext_vector_type(4)))  float f32x4;

#define B_  64
#define T_  512
#define D_  512
#define U_  512
#define NWG 64     // workgroups in recurrent kernel (4 m-tiles x 16 u-slices)

// workspace layout (bytes)
static constexpr size_t OFF_SYNC = 0;                                   // 256 B
static constexpr size_t OFF_WT   = 256;                                 // 3*512*1024*2 = 3145728
static constexpr size_t OFF_HFP  = OFF_WT  + 3145728;                   // 64*512*4 = 131072
static constexpr size_t OFF_HBF  = OFF_HFP + 131072;                    // 64*512*2 = 65536
static constexpr size_t OFF_HR   = OFF_HBF + 65536;                     // 64*512*2 = 65536
static constexpr size_t OFF_GX   = OFF_HR  + 65536;                     // 3*512*64*512*4 = 201326592

union AB { bf16x16 v; bf16x8 h[2]; };

__device__ __forceinline__ f32x8 wmma_bf16(bf16x16 a, bf16x16 b, f32x8 c) {
  // D = A(16x32 bf16) x B(32x16 bf16) + C(16x16 f32)
  return __builtin_amdgcn_wmma_f32_16x16x32_bf16(
      /*neg_a=*/false, a, /*neg_b=*/false, b,
      /*c_mod=*/(short)0, c, /*reuse_a=*/false, /*reuse_b=*/false);
}

__device__ __forceinline__ float hsig(float x) {
  return fminf(fmaxf(0.2f * x + 0.5f, 0.0f), 1.0f);
}

// ---------------------------------------------------------------------------
// Kernel 1a: wT[g][u][k] = bf16(w_g[k][u]),  k in [0,1024): k<512 = h-part
// (state is concatenated FIRST in the reference), k>=512 = x-part.
// ---------------------------------------------------------------------------
__global__ void prep_weights(const float* __restrict__ wr,
                             const float* __restrict__ wz,
                             const float* __restrict__ wh,
                             bf16* __restrict__ wT) {
  int idx = blockIdx.x * 256 + threadIdx.x;   // 3*1024*512 total
  int g = idx >> 19;
  int r = idx & ((1 << 19) - 1);
  int k = r >> 9;
  int u = r & 511;
  const float* w = (g == 0) ? wr : (g == 1) ? wz : wh;
  wT[((size_t)(g * 512 + u)) * 1024 + k] = (bf16)w[(size_t)k * 512 + u];
}

// Kernel 1b: seed h state (fp32 + bf16 mirrors) from h0.
__global__ void init_h(const float* __restrict__ h0,
                       float* __restrict__ hfp, bf16* __restrict__ hbf) {
  int i = blockIdx.x * 256 + threadIdx.x;     // 32768
  float v = h0[i];
  hfp[i] = v;
  hbf[i] = (bf16)v;
}

// ---------------------------------------------------------------------------
// Kernel 2: Gx[g][t][b][u] = bias_g[u] + sum_d x[b][t][d]*w_g[512+d][u]
// One big GEMM: M = B*T = 32768 (m = b*512+t), K = 512, N = 512, per gate.
// Block = 256 threads = 8 waves (4 M x 2 N), tile 64x64, K-chunks of 32.
// Each wave: 16x32 output = two 16x16 WMMA accumulators (N-subtiles +0, +32),
// A operand registers shared by both WMMAs.
// ---------------------------------------------------------------------------
__global__ void xproj(const float* __restrict__ x, const bf16* __restrict__ wT,
                      const float* __restrict__ br, const float* __restrict__ bz,
                      const float* __restrict__ bh, float* __restrict__ gx) {
  const int nTiles = U_ / 64;          // 8
  const int mTiles = (B_ * T_) / 64;   // 512
  int bid = blockIdx.x;
  int g = bid / (mTiles * nTiles);
  int rem = bid % (mTiles * nTiles);
  int mt = rem / nTiles, nt = rem % nTiles;
  int m0 = mt * 64, u0 = nt * 64;
  const float* bias = (g == 0) ? br : (g == 1) ? bz : bh;

  __shared__ alignas(32) bf16 As[64][40];  // 64 m-rows x 32 K, pad to 40 (80B pitch)
  __shared__ alignas(32) bf16 Bs[64][40];  // 64 u-rows x 32 K (transposed W tile)

  int tid = threadIdx.x;
  int wid = tid >> 5, lane = tid & 31;
  int wm = wid & 3, wn = wid >> 2;
  int la = lane & 15, kh = lane >> 4;

  int srow = tid >> 2;            // staging: 4 threads x 8 elems per 32-K row
  int sk8 = (tid & 3) * 8;
  const float* asrc = x + (size_t)(m0 + srow) * D_ + sk8;
  const bf16* bsrc = wT + ((size_t)(g * 512 + u0 + srow)) * 1024 + 512 + sk8;

  f32x8 c0 = {}, c1 = {};
  for (int kc = 0; kc < D_ / 32; ++kc) {
    {  // stage A: 64x32 fp32 -> bf16
      const float* src = asrc + kc * 32;
      f32x4 v0 = *(const f32x4*)src;
      f32x4 v1 = *(const f32x4*)(src + 4);
      bf16x8 o;
      o[0] = (bf16)v0[0]; o[1] = (bf16)v0[1]; o[2] = (bf16)v0[2]; o[3] = (bf16)v0[3];
      o[4] = (bf16)v1[0]; o[5] = (bf16)v1[1]; o[6] = (bf16)v1[2]; o[7] = (bf16)v1[3];
      *(bf16x8*)&As[srow][sk8] = o;
    }
    {  // stage B: 64x32 bf16 (x-part of wT)
      *(bf16x8*)&Bs[srow][sk8] = *(const bf16x8*)(bsrc + kc * 32);
    }
    if (kc + 1 < D_ / 32) {  // warm caches for next K-chunk (global_prefetch_b8)
      __builtin_prefetch(asrc + (kc + 1) * 32, 0, 3);
      __builtin_prefetch(bsrc + (kc + 1) * 32, 0, 3);
    }
    __syncthreads();
    AB a, b0, b1;
    // A layout: row=la, lanes 0-15 K=0..15, lanes 16-31 K=16..31 (contiguous)
    a.h[0] = *(const bf16x8*)&As[wm * 16 + la][kh * 16];
    a.h[1] = *(const bf16x8*)&As[wm * 16 + la][kh * 16 + 8];
    // B layout: col=la, lanes 0-15 K=0..15, lanes 16-31 K=16..31 (contiguous)
    b0.h[0] = *(const bf16x8*)&Bs[wn * 16 + la][kh * 16];
    b0.h[1] = *(const bf16x8*)&Bs[wn * 16 + la][kh * 16 + 8];
    b1.h[0] = *(const bf16x8*)&Bs[32 + wn * 16 + la][kh * 16];
    b1.h[1] = *(const bf16x8*)&Bs[32 + wn * 16 + la][kh * 16 + 8];
    c0 = wmma_bf16(a.v, b0.v, c0);
    c1 = wmma_bf16(a.v, b1.v, c1);
    __syncthreads();
  }

  int ua = u0 + wn * 16 + la;
  int ub = ua + 32;
  float bva = bias[ua];
  float bvb = bias[ub];
  for (int j = 0; j < 8; ++j) {
    int m = m0 + wm * 16 + kh * 8 + j;      // C/D: vgpr j -> M = j + 8*(lane>=16)
    int b_ = m >> 9, t = m & 511;
    size_t base = (((size_t)g * T_ + t) * B_ + b_) * U_;
    gx[base + ua] = c0[j] + bva;
    gx[base + ub] = c1[j] + bvb;
  }
}

// ---------------------------------------------------------------------------
// Kernel 3: persistent recurrence. 64 WGs x 64 threads (2 waves).
// WG (mt, ns) owns h rows [mt*16, +16) and cols [ns*32, +32).
// Grid sync: monotonic atomic counter (zeroed per launch by hipMemsetAsync).
// ---------------------------------------------------------------------------
__device__ __forceinline__ void gsync(unsigned* cnt, unsigned target) {
  __threadfence();
  __syncthreads();
  if (threadIdx.x == 0) {
    __hip_atomic_fetch_add(cnt, 1u, __ATOMIC_ACQ_REL, __HIP_MEMORY_SCOPE_AGENT);
    while (__hip_atomic_load(cnt, __ATOMIC_ACQUIRE, __HIP_MEMORY_SCOPE_AGENT) < target) {
      __builtin_amdgcn_s_sleep(4);
    }
  }
  __syncthreads();
  __threadfence();
}

__global__ void gru_recurrent(const bf16* __restrict__ wT,
                              const float* __restrict__ gx,
                              float* __restrict__ hfp, bf16* __restrict__ hbf,
                              bf16* __restrict__ hr, unsigned* __restrict__ sync,
                              float* __restrict__ out) {
  int wg = blockIdx.x;
  int mt = wg >> 4, ns = wg & 15;
  int m0 = mt * 16, u0 = ns * 32;
  int tid = threadIdx.x;
  int wv = tid >> 5, lane = tid & 31;
  int la = lane & 15, kh = lane >> 4;
  int u = u0 + wv * 16 + la;

  const bf16* Wr = wT;                       // h-part lives at k in [0,512)
  const bf16* Wz = wT + (size_t)512 * 1024;
  const bf16* Wh = wT + (size_t)1024 * 1024;

  const bf16* arowA = hbf + (size_t)(m0 + la) * U_;   // A operand row (b = m0+la)
  const bf16* arowB = hr  + (size_t)(m0 + la) * U_;
  const bf16* browR = Wr + (size_t)(u0 + wv * 16 + la) * 1024;
  const bf16* browZ = Wz + (size_t)(u0 + wv * 16 + la) * 1024;
  const bf16* browH = Wh + (size_t)(u0 + wv * 16 + la) * 1024;

  unsigned nsync = 0;

  for (int t = 0; t < T_; ++t) {
    // ---- phase A: r, z gates; publish h*r (bf16) ----
    f32x8 cr = {}, cz = {};
    for (int kc = 0; kc < U_ / 32; ++kc) {
      AB a, br_, bz_;
      a.h[0]  = *(const bf16x8*)(arowA + kc * 32 + kh * 16);
      a.h[1]  = *(const bf16x8*)(arowA + kc * 32 + kh * 16 + 8);
      br_.h[0] = *(const bf16x8*)(browR + kc * 32 + kh * 16);
      br_.h[1] = *(const bf16x8*)(browR + kc * 32 + kh * 16 + 8);
      bz_.h[0] = *(const bf16x8*)(browZ + kc * 32 + kh * 16);
      bz_.h[1] = *(const bf16x8*)(browZ + kc * 32 + kh * 16 + 8);
      cr = wmma_bf16(a.v, br_.v, cr);
      cz = wmma_bf16(a.v, bz_.v, cz);
    }
    float zreg[8];
    for (int j = 0; j < 8; ++j) {
      int b_ = m0 + kh * 8 + j;
      float gr = gx[(((size_t)0 * T_ + t) * B_ + b_) * U_ + u];
      float gz = gx[(((size_t)1 * T_ + t) * B_ + b_) * U_ + u];
      float rt = hsig(cr[j] + gr);
      zreg[j]  = hsig(cz[j] + gz);
      size_t off = (size_t)b_ * U_ + u;
      hr[off] = (bf16)(hfp[off] * rt);
    }
    gsync(sync, (++nsync) * NWG);

    // ---- phase B: candidate + state update ----
    f32x8 ch = {};
    for (int kc = 0; kc < U_ / 32; ++kc) {
      AB a, bh_;
      a.h[0]  = *(const bf16x8*)(arowB + kc * 32 + kh * 16);
      a.h[1]  = *(const bf16x8*)(arowB + kc * 32 + kh * 16 + 8);
      bh_.h[0] = *(const bf16x8*)(browH + kc * 32 + kh * 16);
      bh_.h[1] = *(const bf16x8*)(browH + kc * 32 + kh * 16 + 8);
      ch = wmma_bf16(a.v, bh_.v, ch);
    }
    for (int j = 0; j < 8; ++j) {
      int b_ = m0 + kh * 8 + j;
      float gh = gx[(((size_t)2 * T_ + t) * B_ + b_) * U_ + u];
      float hh = tanhf(ch[j] + gh);
      float z = zreg[j];
      size_t off = (size_t)b_ * U_ + u;
      float hn = hfp[off] * (1.0f - z) + z * hh;
      hfp[off] = hn;
      hbf[off] = (bf16)hn;
      if (t == T_ - 1) out[off] = hn;
    }
    gsync(sync, (++nsync) * NWG);
  }
}

// ---------------------------------------------------------------------------
extern "C" void kernel_launch(void* const* d_in, const int* in_sizes, int n_in,
                              void* d_out, int out_size, void* d_ws, size_t ws_size,
                              hipStream_t stream) {
  const float* x  = (const float*)d_in[0];
  const float* h0 = (const float*)d_in[1];
  const float* wr = (const float*)d_in[2];
  const float* wz = (const float*)d_in[3];
  const float* wh = (const float*)d_in[4];
  const float* br = (const float*)d_in[5];
  const float* bz = (const float*)d_in[6];
  const float* bh = (const float*)d_in[7];

  char* ws = (char*)d_ws;
  unsigned* sync = (unsigned*)(ws + OFF_SYNC);
  bf16* wT   = (bf16*)(ws + OFF_WT);
  float* hfp = (float*)(ws + OFF_HFP);
  bf16* hbf  = (bf16*)(ws + OFF_HBF);
  bf16* hr   = (bf16*)(ws + OFF_HR);
  float* gx  = (float*)(ws + OFF_GX);
  float* out = (float*)d_out;

  // zero the grid-sync counter every launch (graph-capture safe)
  hipMemsetAsync(sync, 0, 256, stream);

  prep_weights<<<6144, 256, 0, stream>>>(wr, wz, wh, wT);
  init_h<<<128, 256, 0, stream>>>(h0, hfp, hbf);
  xproj<<<3 * 512 * 8, 256, 0, stream>>>(x, wT, br, bz, bh, gx);
  gru_recurrent<<<NWG, 64, 0, stream>>>(wT, gx, hfp, hbf, hr, sync, out);
}
